// ResNeXtHybrid_45191645889350
// MI455X (gfx1250) — matrix operand order repaired and checked by reading, not compiled, not run on hardware
//
#include <hip/hip_runtime.h>
#include <hip/hip_bf16.h>

#define BATCH 65536
#define DIN   256
#define HID   512
#define NG    4
#define GW    128
#define ND    4
#define NL    3
#define BT    64   // batch rows per block

typedef __attribute__((ext_vector_type(16))) __bf16 v16bf;
typedef __attribute__((ext_vector_type(8)))  float  v8f;

union FragU { v16bf v; uint4 q[2]; };

// A-fragment (16x32 bf16, M across lanes 0..15, K striped per ISA layout):
// lane<16: K = {0..7, 16..23}; lane>=16: K = {8..15, 24..31}
__device__ __forceinline__ v16bf load_a_frag(const __bf16* base, int stride, int hf, int l16) {
    FragU f;
    const __bf16* rp = base + l16 * stride + (hf ? 8 : 0);
    f.q[0] = *(const uint4*)(rp);
    f.q[1] = *(const uint4*)(rp + 16);
    return f.v;
}

// B-fragment (32x16 bf16, N across lanes, K striped like A; lane-half selects K+8).
// wt is N-major (element (n,k) at n*K + k); caller passes pointer to (n, k0).
__device__ __forceinline__ v16bf load_b_frag(const __bf16* p) {
    FragU f;
    f.q[0] = *(const uint4*)(p);
    f.q[1] = *(const uint4*)(p + 16);
    return f.v;
}

__global__ void convert_weights(const float* __restrict__ W0, const float* __restrict__ Wg,
                                __bf16* __restrict__ W0T, __bf16* __restrict__ WgT)
{
    long i = (long)blockIdx.x * blockDim.x + threadIdx.x;
    const long n0t = (long)DIN * HID;            // 131072
    const long ngt = (long)NL * NG * ND * GW * GW; // 786432
    if (i < n0t) {
        long n = i / DIN, k = i % DIN;
        W0T[i] = (__bf16)W0[k * HID + n];
    } else if (i < n0t + ngt) {
        long j = i - n0t;
        long blk = j / (GW * GW), r = j % (GW * GW);
        long n = r / GW, k = r % GW;
        WgT[j] = (__bf16)Wg[blk * GW * GW + k * GW + n];
    }
}

__global__ void __launch_bounds__(256)
fused_resnext(const float* __restrict__ x, const __bf16* __restrict__ W0T,
              const float* __restrict__ b0, const __bf16* __restrict__ WgT,
              const float* __restrict__ bg, float* __restrict__ out)
{
    extern __shared__ char smem[];
    __bf16* hbuf = (__bf16*)smem;                             // [BT][HID]  64 KB (x0 tile)
    __bf16* xs   = (__bf16*)(smem + BT * HID * 2);            // [BT][DIN]  32 KB (x stage)
    __bf16* scr  = (__bf16*)(smem + BT * HID * 2 + BT * DIN * 2); // 8 x [16][GW] 32 KB bounce

    const int tid  = threadIdx.x;
    const int w    = tid >> 5;
    const int lane = tid & 31;
    const int hf   = lane >> 4;
    const int l16  = lane & 15;
    const long row0 = (long)blockIdx.x * BT;

    // ---- stage x tile, f32 -> bf16 (coalesced) ----
    const float4* xsrc = (const float4*)(x + row0 * DIN);
    for (int i = tid; i < BT * DIN / 4; i += 256) {
        const float4 v = xsrc[i];
        __bf16* dst = xs + i * 4;
        dst[0] = (__bf16)v.x; dst[1] = (__bf16)v.y;
        dst[2] = (__bf16)v.z; dst[3] = (__bf16)v.w;
    }
    __syncthreads();

    // ---- Phase A: x0 = relu(x @ W0 + b0) -> hbuf (bf16) ----
    {
        const int mt = w & 3;                       // 16-row slab
        const __bf16* abase = xs + (mt * 16) * DIN;
        const int ntBase = (w >> 2) * 16;           // half of the 32 N-tiles
        for (int nt = ntBase; nt < ntBase + 16; ++nt) {
            const int n0 = nt * 16;
            v8f acc = {};
            #pragma unroll
            for (int kc = 0; kc < DIN / 32; ++kc) {
                v16bf a = load_a_frag(abase + kc * 32, DIN, hf, l16);
                v16bf b = load_b_frag(W0T + (long)(n0 + l16) * DIN + kc * 32 + hf * 8);
                acc = __builtin_amdgcn_wmma_f32_16x16x32_bf16(
                        false, a, false, b, (short)0, acc, false, false);
            }
            const float bias = b0[n0 + l16];
            __bf16* hb = hbuf + (mt * 16 + hf * 8) * HID + n0 + l16;
            #pragma unroll
            for (int j = 0; j < 8; ++j) {
                float v = acc[j] + bias;
                hb[j * HID] = (__bf16)(v > 0.f ? v : 0.f);
            }
        }
    }
    __syncthreads();

    // ---- Phase B: grouped layer chain, row-local per (16-row, group) task ----
    const int mt = w & 3;
    __bf16* myscr = scr + w * (16 * GW);
    for (int t = 0; t < 2; ++t) {
        const int g = (w >> 2) * 2 + t;
        const __bf16* hb0 = hbuf + (mt * 16) * HID + g * GW;
        v16bf x0f[4], h[4];
        #pragma unroll
        for (int kc = 0; kc < 4; ++kc) {
            x0f[kc] = load_a_frag(hb0 + kc * 32, HID, hf, l16);
            h[kc] = x0f[kc];
        }
        for (int l = 0; l < NL; ++l) {
            v16bf resid[4];
            #pragma unroll
            for (int kc = 0; kc < 4; ++kc) resid[kc] = h[kc];
            for (int d = 0; d < ND; ++d) {
                const __bf16* wblk = WgT + (long)((l * NG + g) * ND + d) * GW * GW;
                const float*  bgp  = bg + ((l * NG + g) * ND + d) * GW;
                v8f acc[8];
                const v8f z = {};
                #pragma unroll
                for (int nt = 0; nt < 8; ++nt) acc[nt] = z;
                #pragma unroll
                for (int kc = 0; kc < 4; ++kc) {
                    #pragma unroll
                    for (int nt = 0; nt < 8; ++nt) {
                        v16bf b = load_b_frag(wblk + (nt * 16 + l16) * GW + kc * 32 + hf * 8);
                        acc[nt] = __builtin_amdgcn_wmma_f32_16x16x32_bf16(
                                    false, h[kc], false, b, (short)0, acc[nt], false, false);
                    }
                }
                // bias + relu, C-layout -> per-wave LDS bounce -> A-layout
                #pragma unroll
                for (int nt = 0; nt < 8; ++nt) {
                    const float bias = bgp[nt * 16 + l16];
                    __bf16* sp = myscr + (hf * 8) * GW + nt * 16 + l16;
                    #pragma unroll
                    for (int j = 0; j < 8; ++j) {
                        float v = acc[nt][j] + bias;
                        sp[j * GW] = (__bf16)(v > 0.f ? v : 0.f);
                    }
                }
                #pragma unroll
                for (int kc = 0; kc < 4; ++kc)
                    h[kc] = load_a_frag(myscr + kc * 32, GW, hf, l16);
            }
            // layer residual: h = relu(h + resid) elementwise on A-frags
            #pragma unroll
            for (int kc = 0; kc < 4; ++kc) {
                #pragma unroll
                for (int e = 0; e < 16; ++e) {
                    float v = (float)h[kc][e] + (float)resid[kc][e];
                    h[kc][e] = (__bf16)(v > 0.f ? v : 0.f);
                }
            }
        }
        // final: out = relu(h + x0), f32, direct from A-layout (two 32B runs/lane)
        const long orow = row0 + mt * 16 + l16;
        float* op = out + orow * HID + g * GW;
        #pragma unroll
        for (int kc = 0; kc < 4; ++kc) {
            float vals[16];
            #pragma unroll
            for (int e = 0; e < 16; ++e) {
                float v = (float)h[kc][e] + (float)x0f[kc][e];
                vals[e] = v > 0.f ? v : 0.f;
            }
            const int c0 = kc * 32 + hf * 8;
            *(float4*)(op + c0)      = make_float4(vals[0], vals[1], vals[2], vals[3]);
            *(float4*)(op + c0 + 4)  = make_float4(vals[4], vals[5], vals[6], vals[7]);
            *(float4*)(op + c0 + 16) = make_float4(vals[8], vals[9], vals[10], vals[11]);
            *(float4*)(op + c0 + 20) = make_float4(vals[12], vals[13], vals[14], vals[15]);
        }
    }
}

extern "C" void kernel_launch(void* const* d_in, const int* in_sizes, int n_in,
                              void* d_out, int out_size, void* d_ws, size_t ws_size,
                              hipStream_t stream) {
    (void)in_sizes; (void)n_in; (void)out_size; (void)ws_size;
    const float* x  = (const float*)d_in[0];
    const float* W0 = (const float*)d_in[1];
    const float* b0 = (const float*)d_in[2];
    const float* Wg = (const float*)d_in[3];
    const float* bg = (const float*)d_in[4];
    float* out = (float*)d_out;

    __bf16* W0T = (__bf16*)d_ws;                 // 512*256 bf16
    __bf16* WgT = W0T + (long)DIN * HID;         // 786432 bf16

    const long tot = (long)DIN * HID + (long)NL * NG * ND * GW * GW;
    convert_weights<<<(int)((tot + 255) / 256), 256, 0, stream>>>(W0, Wg, W0T, WgT);

    const size_t lds = (size_t)BT * HID * 2 + (size_t)BT * DIN * 2 + (size_t)8 * 16 * GW * 2; // 128 KB
    fused_resnext<<<BATCH / BT, 256, lds, stream>>>(x, W0T, b0, WgT, bg, out);
}
